// PackedLinear_85950885528451
// MI455X (gfx1250) — matrix operand order repaired and checked by reading, not compiled, not run on hardware
//
#include <hip/hip_runtime.h>

// ---- problem constants (from reference) ----
#define M_DIM 2048
#define N_DIM 8192
#define K_DIM 8192
#define NUM_CB_ELEMS (N_DIM * (K_DIM / 128) * 8)   // 4,194,304 codebook entries

// ---- tiling ----
#define BM 128
#define BN 128
#define BK 64
#define LDA 72            // padded LDS row stride (bf16); 144B rows, 16B aligned
#define KSTEPS (K_DIM / BK)

typedef __attribute__((ext_vector_type(16))) __bf16 v16bf;
typedef __attribute__((ext_vector_type(8)))  __bf16 v8bf;
typedef __attribute__((ext_vector_type(4)))  __bf16 v4bf;
typedef __attribute__((ext_vector_type(8)))  float  v8f;

static __device__ __forceinline__ __bf16 f32_to_bf16(float f) {
  union { float f; unsigned u; } v; v.f = f;
  unsigned r = v.u + 0x7FFFu + ((v.u >> 16) & 1u);   // round-to-nearest-even
  return __builtin_bit_cast(__bf16, (unsigned short)(r >> 16));
}

static __device__ __forceinline__ v16bf ld_frag(const __bf16* p0, const __bf16* p1) {
  v8bf lo = *(const v8bf*)p0;   // ds_load_b128
  v8bf hi = *(const v8bf*)p1;   // ds_load_b128
  return __builtin_shufflevector(lo, hi, 0,1,2,3,4,5,6,7,8,9,10,11,12,13,14,15);
}

// 16B global -> LDS async copy (ASYNCcnt-tracked). VDST VGPR carries the LDS
// byte address, VADDR pair carries the 64-bit global address (GV mode).
static __device__ __forceinline__ void async_b128(const void* g, unsigned lds_byte) {
  asm volatile("global_load_async_to_lds_b128 %0, %1, off"
               :: "v"(lds_byte), "v"(g) : "memory");
}
static __device__ __forceinline__ void wait_async0() {
#if __has_builtin(__builtin_amdgcn_s_wait_asynccnt)
  __builtin_amdgcn_s_wait_asynccnt(0);
#else
  asm volatile("s_wait_asynccnt 0x0" ::: "memory");
#endif
}

// ---- one-time fp32 -> bf16 stream conversion (x and codebook) ----
__global__ __launch_bounds__(256)
void convert_f32_bf16(const float* __restrict__ s, __bf16* __restrict__ d) {
  long i = ((long)blockIdx.x * 256 + threadIdx.x) * 4;
  const float4 v = *(const float4*)(s + i);
  v4bf h;
  h[0] = f32_to_bf16(v.x); h[1] = f32_to_bf16(v.y);
  h[2] = f32_to_bf16(v.z); h[3] = f32_to_bf16(v.w);
  *(v4bf*)(d + i) = h;
}

// PRE = true : x & codebook pre-converted to bf16 in workspace; X tiles staged
//              via global_load_async_to_lds_b128, decode gathers bf16 directly.
// PRE = false: fully fused fallback (fp32 sources, convert while staging).
template <bool PRE>
__global__ __launch_bounds__(256)
void packed_linear_wmma(const float*  __restrict__ xf,
                        const __bf16* __restrict__ xb,
                        const int*    __restrict__ packed,
                        const float*  __restrict__ codebook,
                        const __bf16* __restrict__ codebook_bf,
                        float*        __restrict__ out) {
  __shared__ __bf16 Xs[2][BM * LDA];
  __shared__ __bf16 Ws[2][BN * LDA];

  const int tid    = threadIdx.x;
  const int lane   = tid & 31;
  const int wave   = tid >> 5;
  const int lane16 = lane & 15;
  const int half   = lane >> 4;
  const int waveM  = wave & 1;    // 2 waves over M -> 64 rows each
  const int waveN  = wave >> 1;   // 4 waves over N -> 32 cols each

  const int m0 = blockIdx.y * BM;
  const int n0 = blockIdx.x * BN;

  // ---- hoisted per-thread addressing (all 32-bit offsets) ----
  unsigned xgOff[4];  unsigned xsAddr[2][4];          // PRE: async X staging
  unsigned xfOff[8];  int      xlOff8[8];             // !PRE: manual X staging
  unsigned pOff[4];   int      cbRow[4];   int wOff[4];

  if constexpr (PRE) {
    #pragma unroll
    for (int i = 0; i < 4; ++i) {
      int idx = tid + i * 256;            // 1024 16B chunks of the X tile
      int row = idx >> 3;                 // tile row (8 chunks per 64-elem row)
      int c   = idx & 7;
      xgOff[i] = (unsigned)(m0 + row) * K_DIM + c * 8;
      int xl = row * LDA + c * 8;
      xsAddr[0][i] = (unsigned)(uintptr_t)&Xs[0][xl];
      xsAddr[1][i] = (unsigned)(uintptr_t)&Xs[1][xl];
    }
  } else {
    #pragma unroll
    for (int i = 0; i < 8; ++i) {
      int idx4 = tid + i * 256;           // 2048 float4s of the X tile
      int ml   = idx4 >> 4;
      int k4   = (idx4 & 15) << 2;
      xfOff[i]  = (unsigned)(m0 + ml) * K_DIM + k4;
      xlOff8[i] = ml * LDA + k4;
    }
  }
  #pragma unroll
  for (int i = 0; i < 4; ++i) {
    int g  = tid + i * 256;               // 1024 groups of 8 weights
    int nl = g >> 3;
    int kg = (g & 7) << 3;
    unsigned rowFlat = (unsigned)(n0 + nl) * K_DIM + kg;
    pOff[i]  = 3u * (rowFlat >> 3);       // advances by +24 ints per K-step
    cbRow[i] = (n0 + nl) * 512;           // n*64 blocks * 8 entries
    wOff[i]  = nl * LDA + kg;
  }

  v8f acc[4][2] = {};

  auto stage = [&](int t, int buf) {
    const int k0 = t * BK;
    if constexpr (PRE) {
      #pragma unroll
      for (int i = 0; i < 4; ++i)
        async_b128(xb + (xgOff[i] + k0), xsAddr[buf][i]);
    } else {
      #pragma unroll
      for (int i = 0; i < 8; ++i) {
        const float4 xv = *(const float4*)(xf + (xfOff[i] + k0));
        v4bf h;
        h[0] = f32_to_bf16(xv.x); h[1] = f32_to_bf16(xv.y);
        h[2] = f32_to_bf16(xv.z); h[3] = f32_to_bf16(xv.w);
        *(v4bf*)&Xs[buf][xlOff8[i]] = h;
      }
    }
    const int cbStep = (k0 >> 7) << 3;    // +8 entries every other K-step
    #pragma unroll
    for (int i = 0; i < 4; ++i) {
      const int* p = packed + (pOff[i] + 24 * t);
      __builtin_prefetch(p + 24, 0, 1);   // next K-step's packed bits
      unsigned bits = (unsigned)(p[0] & 0xFF)
                    | ((unsigned)(p[1] & 0xFF) << 8)
                    | ((unsigned)(p[2] & 0xFF) << 16);
      v8bf w;
      if constexpr (PRE) {
        const __bf16* cb = codebook_bf + cbRow[i] + cbStep;
        #pragma unroll
        for (int j = 0; j < 8; ++j)
          w[j] = cb[(bits >> (3 * j)) & 7];
      } else {
        const float* cb = codebook + cbRow[i] + cbStep;
        #pragma unroll
        for (int j = 0; j < 8; ++j)
          w[j] = f32_to_bf16(cb[(bits >> (3 * j)) & 7]);
      }
      *(v8bf*)&Ws[buf][wOff[i]] = w;
    }
  };

  auto compute = [&](int buf) {
    #pragma unroll
    for (int ks = 0; ks < BK; ks += 32) {
      v16bf a[4], b[2];
      #pragma unroll
      for (int mt = 0; mt < 4; ++mt) {
        const __bf16* base = &Xs[buf][(waveM * 64 + mt * 16 + lane16) * LDA + ks + half * 8];
        a[mt] = ld_frag(base, base + 16);
      }
      #pragma unroll
      for (int nt = 0; nt < 2; ++nt) {
        const __bf16* base = &Ws[buf][(waveN * 32 + nt * 16 + lane16) * LDA + ks + half * 8];
        b[nt] = ld_frag(base, base + 16);
      }
      #pragma unroll
      for (int mt = 0; mt < 4; ++mt)
        #pragma unroll
        for (int nt = 0; nt < 2; ++nt)
          acc[mt][nt] = __builtin_amdgcn_wmma_f32_16x16x32_bf16(
              false, a[mt], false, b[nt], (short)0, acc[mt][nt], false, false);
    }
  };

  // ---- software-pipelined main loop: stage(t+1) overlaps compute(t) ----
  stage(0, 0);
  if constexpr (PRE) wait_async0();
  __syncthreads();
  for (int t = 0; t < KSTEPS; ++t) {
    const int cur = t & 1;
    if (t + 1 < KSTEPS) stage(t + 1, cur ^ 1);
    compute(cur);
    if constexpr (PRE) wait_async0();
    __syncthreads();
  }

  // ---- epilogue: C/D f32 16x16 layout -> out[m][n] ----
  #pragma unroll
  for (int mt = 0; mt < 4; ++mt)
    #pragma unroll
    for (int nt = 0; nt < 2; ++nt)
      #pragma unroll
      for (int r = 0; r < 8; ++r) {
        int m = m0 + waveM * 64 + mt * 16 + half * 8 + r;
        int n = n0 + waveN * 32 + nt * 16 + lane16;
        out[(long)m * N_DIM + n] = acc[mt][nt][r];
      }
}

extern "C" void kernel_launch(void* const* d_in, const int* in_sizes, int n_in,
                              void* d_out, int out_size, void* d_ws, size_t ws_size,
                              hipStream_t stream) {
  (void)in_sizes; (void)n_in; (void)out_size;
  const float* x        = (const float*)d_in[0];
  const int*   packed   = (const int*)d_in[1];
  const float* codebook = (const float*)d_in[2];
  float*       out      = (float*)d_out;

  const size_t xb_bytes = (size_t)M_DIM * K_DIM * 2;       // 32 MB
  const size_t cb_bytes = (size_t)NUM_CB_ELEMS * 2;        // 8 MB
  dim3 grid(N_DIM / BN, M_DIM / BM);                       // 64 x 16 workgroups

  if (ws_size >= xb_bytes + cb_bytes) {
    __bf16* xb  = (__bf16*)d_ws;
    __bf16* cbb = (__bf16*)((char*)d_ws + xb_bytes);
    convert_f32_bf16<<<(M_DIM * (long)K_DIM) / 1024, 256, 0, stream>>>(x, xb);
    convert_f32_bf16<<<NUM_CB_ELEMS / 1024, 256, 0, stream>>>(codebook, cbb);
    packed_linear_wmma<true><<<grid, dim3(256), 0, stream>>>(
        nullptr, xb, packed, codebook, cbb, out);
  } else {
    packed_linear_wmma<false><<<grid, dim3(256), 0, stream>>>(
        x, nullptr, packed, codebook, nullptr, out);
  }
}
